// AttentionEncoder_42545946034846
// MI455X (gfx1250) — matrix operand order repaired and checked by reading, not compiled, not run on hardware
//
#include <hip/hip_runtime.h>
#include <cstdint>
#include <cstddef>

// ---------------------------------------------------------------------------
// Types
// ---------------------------------------------------------------------------
typedef __bf16 bf16;
typedef __attribute__((ext_vector_type(16))) __bf16 v16bf;
typedef __attribute__((ext_vector_type(8)))  __bf16 bf16x8;
typedef __attribute__((ext_vector_type(8)))  float  v8f;
typedef __attribute__((ext_vector_type(4)))  int    v4i;

__device__ __forceinline__ bf16 f2bf(float f) {
  union { float f; uint32_t u; } c; c.f = f;
  uint32_t u = c.u;
  uint32_t r = u + 0x7FFFu + ((u >> 16) & 1u);   // round-to-nearest-even
  uint16_t h = (uint16_t)(r >> 16);
  return __builtin_bit_cast(bf16, h);
}

// Model constants (B=2, S=2048, D=1024, FF=4096, H=16, DH=64, L=4)
#define MROWS 4096          // B*S
#define DMODEL 1024
#define DFF   4096
#define NHEAD 16
#define DHEAD 64
#define SEQ   2048
#define NLAYER 4

// ---------------------------------------------------------------------------
// CDNA5 async global->LDS staging (ASYNCcnt path), guarded so the build
// degrades to a manual copy if the builtin is absent on this toolchain.
// Builtin signature (from hipcc diagnostic): arg0 = v4i* in global (AS1),
// arg1 = v4i* in LDS (AS3), arg2 = imm offset, arg3 = imm cpol.
// ---------------------------------------------------------------------------
#if __has_builtin(__builtin_amdgcn_global_load_async_to_lds_b128)
#define HAVE_ASYNC_LDS 1
typedef __attribute__((address_space(1))) v4i* gv4p;
typedef __attribute__((address_space(3))) v4i* lv4p;
__device__ __forceinline__ void async_copy16(const void* g, void* lds) {
  // Generic LDS addresses carry the byte offset in the low 32 bits.
  gv4p gp = (gv4p)(uintptr_t)g;
  lv4p lp = (lv4p)(uint32_t)(uintptr_t)lds;
  __builtin_amdgcn_global_load_async_to_lds_b128(gp, lp, 0, 0);
}
#else
#define HAVE_ASYNC_LDS 0
#endif

__device__ __forceinline__ void wait_async0() {
#if __has_builtin(__builtin_amdgcn_s_wait_asynccnt)
  __builtin_amdgcn_s_wait_asynccnt(0);
#else
  asm volatile("s_wait_asynccnt 0" ::: "memory");
#endif
}

// ---------------------------------------------------------------------------
// fp32 -> bf16 elementwise convert
// ---------------------------------------------------------------------------
__global__ __launch_bounds__(256)
void f32_to_bf16_kernel(const float* __restrict__ in, bf16* __restrict__ out, int n) {
  int i = blockIdx.x * 256 + threadIdx.x;
  if (i < n) out[i] = f2bf(in[i]);
}

// ---------------------------------------------------------------------------
// V transpose per (b,h): v[(b*S+s)*D + h*64 + d] -> vt[((b*16+h)*64 + d)*S + s]
// ---------------------------------------------------------------------------
__global__ __launch_bounds__(256)
void transpose_v_kernel(const bf16* __restrict__ v, bf16* __restrict__ vt) {
  int i = blockIdx.x * 256 + threadIdx.x;
  if (i >= MROWS * DMODEL) return;
  int d = i & 63;
  int h = (i >> 6) & 15;
  int s = (i >> 10) & (SEQ - 1);
  int b = i >> 21;
  vt[((size_t)((b * 16 + h) * 64 + d)) * SEQ + s] = v[i];
}

// ---------------------------------------------------------------------------
// bf16 GEMM:  C[M,N] = A[M,K](bf16) * W[K,N](f32, converted in-flight) + bias
// FLAGS: bit0 = ReLU, bit1 = write f32 out, bit2 = write bf16 out
// Block = 128 threads (4 waves). Block tile 128x64, wave tile 32x64, K-step 32.
// Per K-step per wave: 2 A-frags, 4 B-frags, 8 v_wmma.
// A tile staged with GLOBAL_LOAD_ASYNC_TO_LDS_B128 when available.
// ---------------------------------------------------------------------------
#define LDAH 40   // padded LDS row stride in halves (80B, 16B aligned)

template <int FLAGS>
__global__ __launch_bounds__(128)
void gemm_bf16_kernel(const bf16* __restrict__ A, const float* __restrict__ W,
                      const float* __restrict__ bias,
                      float* __restrict__ outF, bf16* __restrict__ outB,
                      int M, int N, int K)
{
  __shared__ bf16 As[128 * LDAH];  // 128 rows x 32 halves (A tile, row-major)
  __shared__ bf16 Bs[64 * LDAH];   // 64 cols x 32 halves (W tile, transposed [n][k])

  const int tid  = threadIdx.x;
  const int wid  = tid >> 5;
  const int lane = tid & 31;
  const int lh   = lane & 15;
  const int hi   = lane >> 4;           // 0 or 1 (half-wave)
  const int mb   = blockIdx.y * 128;
  const int nb   = blockIdx.x * 64;

  v8f acc[2][4] = {};

  for (int k0 = 0; k0 < K; k0 += 32) {
    // Stage A tile: 128x32 bf16 = 512 16-byte chunks, 4 per thread
    #pragma unroll
    for (int it = 0; it < 4; ++it) {
      int idx = tid + it * 128;
      int r  = idx >> 2;
      int c8 = (idx & 3) << 3;
      const bf16* src = A + (size_t)(mb + r) * K + k0 + c8;
      bf16* dst = &As[r * LDAH + c8];
#if HAVE_ASYNC_LDS
      async_copy16(src, dst);
#else
      *(bf16x8*)dst = *(const bf16x8*)src;
#endif
    }
    // Stage W tile: 32x64 f32 -> convert -> transposed bf16 [n][k]
    #pragma unroll
    for (int it = 0; it < 4; ++it) {
      int idx = tid + it * 128;                 // 512 chunks of 4 floats
      int kk = idx >> 4;
      int n4 = (idx & 15) << 2;
      float4 wv = *(const float4*)(W + (size_t)(k0 + kk) * N + nb + n4);
      Bs[(n4 + 0) * LDAH + kk] = f2bf(wv.x);
      Bs[(n4 + 1) * LDAH + kk] = f2bf(wv.y);
      Bs[(n4 + 2) * LDAH + kk] = f2bf(wv.z);
      Bs[(n4 + 3) * LDAH + kk] = f2bf(wv.w);
    }
#if HAVE_ASYNC_LDS
    wait_async0();
#endif
    __syncthreads();

    // A fragments (16x32 each): lane<16 row lh K{0..7,16..23}; lane>=16 K{8..15,24..31}
    union Frag { v16bf v; bf16x8 x[2]; };
    Frag af[2];
    #pragma unroll
    for (int mf = 0; mf < 2; ++mf) {
      const bf16* ap = &As[(wid * 32 + mf * 16 + lh) * LDAH + hi * 8];
      af[mf].x[0] = *(const bf16x8*)(ap);
      af[mf].x[1] = *(const bf16x8*)(ap + 16);
    }
    #pragma unroll
    for (int j = 0; j < 4; ++j) {
      // B fragment (32x16): lane holds column (j*16+lh), lane<16 K{0..7,16..23}; lane>=16 K{8..15,24..31}
      Frag bfr;
      const bf16* bp = &Bs[(j * 16 + lh) * LDAH + hi * 8];
      bfr.x[0] = *(const bf16x8*)(bp);
      bfr.x[1] = *(const bf16x8*)(bp + 16);
      acc[0][j] = __builtin_amdgcn_wmma_f32_16x16x32_bf16(
          false, af[0].v, false, bfr.v, (short)0, acc[0][j], false, false);
      acc[1][j] = __builtin_amdgcn_wmma_f32_16x16x32_bf16(
          false, af[1].v, false, bfr.v, (short)0, acc[1][j], false, false);
    }
    __syncthreads();
  }

  // Epilogue: bias + optional ReLU, compile-time selected stores
  #pragma unroll
  for (int j = 0; j < 4; ++j) {
    int col = nb + j * 16 + lh;
    float bv = bias[col];
    #pragma unroll
    for (int mf = 0; mf < 2; ++mf) {
      #pragma unroll
      for (int r = 0; r < 8; ++r) {
        int row = mb + wid * 32 + mf * 16 + r + hi * 8;
        float v = acc[mf][j][r] + bv;
        if (FLAGS & 1) v = fmaxf(v, 0.0f);
        size_t o = (size_t)row * N + col;
        if (FLAGS & 2) outF[o] = v;
        if (FLAGS & 4) outB[o] = f2bf(v);
      }
    }
  }
}

// ---------------------------------------------------------------------------
// Flash-attention: one wave per (b, h, 16-query tile). Online softmax.
// ---------------------------------------------------------------------------
__global__ __launch_bounds__(128)
void flash_attn_kernel(const bf16* __restrict__ q, const bf16* __restrict__ k,
                       const bf16* __restrict__ vt, bf16* __restrict__ ctx)
{
  __shared__ bf16 Pl[4][16 * 32];   // per-wave P tile (row stride 32 halves)

  const int tid  = threadIdx.x;
  const int wid  = tid >> 5;
  const int lane = tid & 31;
  const int lh   = lane & 15;
  const int hi   = lane >> 4;

  int task = blockIdx.x * 4 + wid;             // 4096 tasks = b(2) x h(16) x qtile(128)
  int i0 = (task & 127) << 4;
  int h  = (task >> 7) & 15;
  int b  = task >> 11;

  union Frag { v16bf v; bf16x8 x[2]; };

  // Q fragments (A operand, 16x32 each; DH=64 -> two fragments)
  Frag aq0, aq1;
  {
    const bf16* p = q + ((size_t)(b * SEQ + i0 + lh)) * DMODEL + h * DHEAD + hi * 8;
    aq0.x[0] = *(const bf16x8*)(p);       aq0.x[1] = *(const bf16x8*)(p + 16);
    aq1.x[0] = *(const bf16x8*)(p + 32);  aq1.x[1] = *(const bf16x8*)(p + 48);
  }

  float mrow[8], srow[8];
  v8f o[4] = {};
  #pragma unroll
  for (int r = 0; r < 8; ++r) { mrow[r] = -1e30f; srow[r] = 0.0f; }
  const float scale = 0.125f;   // 1/sqrt(DH)

  for (int j0 = 0; j0 < SEQ; j0 += 32) {
    // --- scores: two 16x16 tiles (keys j0..j0+15, j0+16..j0+31) ---
    v8f sc[2];
    #pragma unroll
    for (int t = 0; t < 2; ++t) {
      const bf16* kp = k + ((size_t)(b * SEQ + j0 + t * 16 + lh)) * DMODEL
                         + h * DHEAD + hi * 8;
      Frag b0, b1;                       // K^T fragments: column=key, inner=d
      b0.x[0] = *(const bf16x8*)(kp);      b0.x[1] = *(const bf16x8*)(kp + 16);
      b1.x[0] = *(const bf16x8*)(kp + 32); b1.x[1] = *(const bf16x8*)(kp + 48);
      v8f z = {};
      z = __builtin_amdgcn_wmma_f32_16x16x32_bf16(false, aq0.v, false, b0.v,
                                                  (short)0, z, false, false);
      z = __builtin_amdgcn_wmma_f32_16x16x32_bf16(false, aq1.v, false, b1.v,
                                                  (short)0, z, false, false);
      sc[t] = z;
    }

    // --- online softmax (row r lives in VGPR r across 16 lanes of this half) ---
    float pl0[8], pl1[8];
    #pragma unroll
    for (int r = 0; r < 8; ++r) {
      float a0 = sc[0][r] * scale, a1 = sc[1][r] * scale;
      float mx = fmaxf(a0, a1);
      mx = fmaxf(mx, __shfl_xor(mx, 1));
      mx = fmaxf(mx, __shfl_xor(mx, 2));
      mx = fmaxf(mx, __shfl_xor(mx, 4));
      mx = fmaxf(mx, __shfl_xor(mx, 8));
      float mi   = fmaxf(mrow[r], mx);
      float corr = __expf(mrow[r] - mi);
      mrow[r] = mi;
      float p0 = __expf(a0 - mi);
      float p1 = __expf(a1 - mi);
      float rs = p0 + p1;
      rs += __shfl_xor(rs, 1); rs += __shfl_xor(rs, 2);
      rs += __shfl_xor(rs, 4); rs += __shfl_xor(rs, 8);
      srow[r] = srow[r] * corr + rs;
      #pragma unroll
      for (int f = 0; f < 4; ++f) o[f][r] *= corr;
      pl0[r] = p0; pl1[r] = p1;
    }

    // --- transpose P (acc layout -> A layout) through per-wave LDS ---
    bf16* pw = &Pl[wid][0];
    #pragma unroll
    for (int r = 0; r < 8; ++r) {
      int row = r + hi * 8;
      pw[row * 32 + lh]      = f2bf(pl0[r]);
      pw[row * 32 + 16 + lh] = f2bf(pl1[r]);
    }
    asm volatile("s_wait_dscnt 0" ::: "memory");
    Frag ap;
    const bf16* pr = pw + lh * 32 + hi * 8;
    ap.x[0] = *(const bf16x8*)(pr);
    ap.x[1] = *(const bf16x8*)(pr + 16);

    // --- O += P (16x32) * V (32x64) via 4 fragments of Vt ---
    #pragma unroll
    for (int f = 0; f < 4; ++f) {
      const bf16* vp = vt + ((size_t)((b * 16 + h) * 64 + f * 16 + lh)) * SEQ
                          + j0 + hi * 8;
      Frag bv;
      bv.x[0] = *(const bf16x8*)(vp);
      bv.x[1] = *(const bf16x8*)(vp + 16);
      o[f] = __builtin_amdgcn_wmma_f32_16x16x32_bf16(
          false, ap.v, false, bv.v, (short)0, o[f], false, false);
    }
  }

  // --- finalize: divide by running sum, scatter to ctx ---
  #pragma unroll
  for (int r = 0; r < 8; ++r) {
    float inv = 1.0f / srow[r];
    int row = i0 + r + hi * 8;
    bf16* cp = ctx + ((size_t)(b * SEQ + row)) * DMODEL + h * DHEAD + lh;
    #pragma unroll
    for (int f = 0; f < 4; ++f) cp[f * 16] = f2bf(o[f][r] * inv);
  }
}

// ---------------------------------------------------------------------------
// Residual + LayerNorm. One row (1024 elems) per 256-thread block.
// Writes f32 (next residual / final output) and bf16 (next GEMM operand).
// ---------------------------------------------------------------------------
__global__ __launch_bounds__(256)
void residual_ln_kernel(const float* __restrict__ y, const float* __restrict__ x,
                        const float* __restrict__ g, const float* __restrict__ beta,
                        float* __restrict__ xoutF, bf16* __restrict__ xoutB)
{
  __shared__ float red[256];
  const int row = blockIdx.x;
  const float* yr = y + (size_t)row * DMODEL;
  const float* xr = x + (size_t)row * DMODEL;

  float vals[4];
  float s = 0.0f;
  #pragma unroll
  for (int i = 0; i < 4; ++i) {
    int c = threadIdx.x + i * 256;
    vals[i] = yr[c] + xr[c];
    s += vals[i];
  }
  red[threadIdx.x] = s; __syncthreads();
  for (int off = 128; off > 0; off >>= 1) {
    if (threadIdx.x < off) red[threadIdx.x] += red[threadIdx.x + off];
    __syncthreads();
  }
  float mean = red[0] * (1.0f / DMODEL);
  __syncthreads();

  float vs = 0.0f;
  #pragma unroll
  for (int i = 0; i < 4; ++i) { float d = vals[i] - mean; vs += d * d; }
  red[threadIdx.x] = vs; __syncthreads();
  for (int off = 128; off > 0; off >>= 1) {
    if (threadIdx.x < off) red[threadIdx.x] += red[threadIdx.x + off];
    __syncthreads();
  }
  float inv = rsqrtf(red[0] * (1.0f / DMODEL) + 1e-6f);

  #pragma unroll
  for (int i = 0; i < 4; ++i) {
    int c = threadIdx.x + i * 256;
    float ov = (vals[i] - mean) * inv * g[c] + beta[c];
    xoutF[(size_t)row * DMODEL + c] = ov;
    xoutB[(size_t)row * DMODEL + c] = f2bf(ov);
  }
}

// ---------------------------------------------------------------------------
// Host launch
// ---------------------------------------------------------------------------
static void launch_gemm(const bf16* A, const float* W, const float* bias,
                        float* outF, bf16* outB, int M, int N, int K, int flags,
                        hipStream_t s) {
  dim3 grid(N / 64, M / 128);
  if (flags == 4)
    gemm_bf16_kernel<4><<<grid, 128, 0, s>>>(A, W, bias, outF, outB, M, N, K);
  else if (flags == 5)
    gemm_bf16_kernel<5><<<grid, 128, 0, s>>>(A, W, bias, outF, outB, M, N, K);
  else
    gemm_bf16_kernel<2><<<grid, 128, 0, s>>>(A, W, bias, outF, outB, M, N, K);
}

extern "C" void kernel_launch(void* const* d_in, const int* in_sizes, int n_in,
                              void* d_out, int out_size, void* d_ws, size_t ws_size,
                              hipStream_t stream) {
  (void)in_sizes; (void)n_in; (void)out_size; (void)ws_size;
  const float* hs   = (const float*)d_in[0];
  const float* Wq   = (const float*)d_in[1];  const float* bq = (const float*)d_in[2];
  const float* Wk   = (const float*)d_in[3];  const float* bk = (const float*)d_in[4];
  const float* Wv   = (const float*)d_in[5];  const float* bv = (const float*)d_in[6];
  const float* Wp   = (const float*)d_in[7];  const float* bp = (const float*)d_in[8];
  const float* ln1g = (const float*)d_in[9];  const float* ln1b = (const float*)d_in[10];
  const float* W1   = (const float*)d_in[11]; const float* b1 = (const float*)d_in[12];
  const float* W2   = (const float*)d_in[13]; const float* b2 = (const float*)d_in[14];
  const float* ln2g = (const float*)d_in[15]; const float* ln2b = (const float*)d_in[16];

  const size_t ACT = (size_t)MROWS * DMODEL;     // 4M elements
  char* ws = (char*)d_ws;
  float* x    = (float*)ws;              ws += ACT * sizeof(float);   // 16 MB
  float* ybuf = (float*)ws;              ws += ACT * sizeof(float);   // 16 MB
  bf16* xbf   = (bf16*)ws;               ws += ACT * sizeof(bf16);    //  8 MB
  bf16* qbf   = (bf16*)ws;               ws += ACT * sizeof(bf16);
  bf16* kbf   = (bf16*)ws;               ws += ACT * sizeof(bf16);
  bf16* vbf   = (bf16*)ws;               ws += ACT * sizeof(bf16);
  bf16* vtb   = (bf16*)ws;               ws += ACT * sizeof(bf16);
  bf16* ctxb  = (bf16*)ws;               ws += ACT * sizeof(bf16);
  bf16* h1bf  = (bf16*)ws;               ws += (size_t)MROWS * DFF * sizeof(bf16); // 32 MB

  // x = hidden_states (fresh every call); xbf = bf16(x)
  (void)hipMemcpyAsync(x, hs, ACT * sizeof(float), hipMemcpyDeviceToDevice, stream);
  f32_to_bf16_kernel<<<(int)((ACT + 255) / 256), 256, 0, stream>>>(x, xbf, (int)ACT);

  for (int l = 0; l < NLAYER; ++l) {
    const size_t wo = (size_t)l * DMODEL * DMODEL;
    const size_t fo = (size_t)l * DMODEL * DFF;
    // Q, K, V projections (bf16 out)
    launch_gemm(xbf, Wq + wo, bq + (size_t)l * DMODEL, nullptr, qbf,
                MROWS, DMODEL, DMODEL, 4, stream);
    launch_gemm(xbf, Wk + wo, bk + (size_t)l * DMODEL, nullptr, kbf,
                MROWS, DMODEL, DMODEL, 4, stream);
    launch_gemm(xbf, Wv + wo, bv + (size_t)l * DMODEL, nullptr, vbf,
                MROWS, DMODEL, DMODEL, 4, stream);
    // V transpose + flash attention
    transpose_v_kernel<<<(int)((ACT + 255) / 256), 256, 0, stream>>>(vbf, vtb);
    flash_attn_kernel<<<1024, 128, 0, stream>>>(qbf, kbf, vtb, ctxb);
    // Output projection (f32 out), residual + LN1 (in-place on x)
    launch_gemm(ctxb, Wp + wo, bp + (size_t)l * DMODEL, ybuf, nullptr,
                MROWS, DMODEL, DMODEL, 2, stream);
    residual_ln_kernel<<<MROWS, 256, 0, stream>>>(
        ybuf, x, ln1g + (size_t)l * DMODEL, ln1b + (size_t)l * DMODEL, x, xbf);
    // FFN: relu(x @ W1 + b1) @ W2 + b2, residual + LN2
    launch_gemm(xbf, W1 + fo, b1 + (size_t)l * DFF, nullptr, h1bf,
                MROWS, DFF, DMODEL, 5, stream);
    launch_gemm(h1bf, W2 + fo, b2 + (size_t)l * DMODEL, ybuf, nullptr,
                MROWS, DMODEL, DFF, 2, stream);
    float* xo = (l == NLAYER - 1) ? (float*)d_out : x;
    residual_ln_kernel<<<MROWS, 256, 0, stream>>>(
        ybuf, x, ln2g + (size_t)l * DMODEL, ln2b + (size_t)l * DMODEL, xo, xbf);
  }
}
